// SelfAttention_63471026700585
// MI455X (gfx1250) — compile-verified, matrix operand order checked by reading, not run on hardware
//
#include <hip/hip_runtime.h>
#include <hip/hip_bf16.h>
#include <cstdint>
#include <math.h>

#ifndef USE_ASYNC_LDS
#define USE_ASYNC_LDS 1
#endif

#define NTOK 4096   // H*W
#define CCH  256    // C
#define CQK  32     // C/8
#define NBAT 4      // B

typedef __bf16 v16bf __attribute__((ext_vector_type(16)));
typedef __bf16 v8bf  __attribute__((ext_vector_type(8)));
typedef float  v8f   __attribute__((ext_vector_type(8)));

static __device__ inline v8f wmma_bf16(v16bf a, v16bf b, v8f c) {
  // D = A(16x32 bf16) * B(32x16 bf16) + C(16x16 f32)
  return __builtin_amdgcn_wmma_f32_16x16x32_bf16(false, a, false, b,
                                                 (short)0, c, false, false);
}

static __device__ inline v16bf join16(v8bf lo, v8bf hi) {
  return __builtin_shufflevector(lo, hi, 0,1,2,3,4,5,6,7,8,9,10,11,12,13,14,15);
}

// Load one A/B operand half-per-lane: 8 bf16 at base, 8 bf16 at base+16.
// Matches the 16-bit 16x32 A / 32x16 B per-lane K ordering:
// lanes 0-15 hold K={0..7,16..23}, lanes 16-31 hold K={8..15,24..31},
// with the kb=hi?8:0 offset folded into `base` by the caller.
static __device__ inline v16bf load_op(const __bf16* base) {
  return join16(*(const v8bf*)base, *(const v8bf*)(base + 16));
}

// ---------------------------------------------------------------------------
// Kernel 0: one-shot f32 -> bf16 conversion of the weight matrices, so the
// 1024 projection waves don't each redo identical conversions.
// ---------------------------------------------------------------------------
__global__ __launch_bounds__(256)
void cvt_weights(const float* __restrict__ Wq, const float* __restrict__ Wk,
                 const float* __restrict__ Wv,
                 __bf16* __restrict__ wqb, __bf16* __restrict__ wkb,
                 __bf16* __restrict__ wvb)
{
  const int i = blockIdx.x * 256 + threadIdx.x;
  if (i < CQK * CCH) { wqb[i] = (__bf16)Wq[i]; wkb[i] = (__bf16)Wk[i]; }
  if (i < CCH * CCH) { wvb[i] = (__bf16)Wv[i]; }
}

// ---------------------------------------------------------------------------
// Kernel 1: Q/K/V projections (1x1 convs) via bf16 WMMA.
//   qT,kT : bf16 [B][N][32]   (transposed so attention tiles are contiguous)
//   vB    : bf16 [B][256][N]
// ---------------------------------------------------------------------------
__global__ __launch_bounds__(256)
void proj_kernel(const float* __restrict__ x,
                 const __bf16* __restrict__ Wq, const float* __restrict__ bq,
                 const __bf16* __restrict__ Wk, const float* __restrict__ bk,
                 const __bf16* __restrict__ Wv, const float* __restrict__ bv,
                 __bf16* __restrict__ qT, __bf16* __restrict__ kT,
                 __bf16* __restrict__ vB)
{
  const int b    = blockIdx.x >> 5;           // 32 blocks per batch
  const int wave = threadIdx.x >> 5;
  const int lane = threadIdx.x & 31;
  const int l16  = lane & 15;
  const int kb   = (lane >= 16) ? 8 : 0;
  const int n0   = ((blockIdx.x & 31) * 8 + wave) * 16;
  const int n    = n0 + l16;

  // Preload the 8 B-operand tiles of x (channel chunks of 32) for this n-tile.
  // Each x element is loaded/converted exactly once across the whole grid.
  v16bf bt[8];
  #pragma unroll
  for (int kc = 0; kc < 8; ++kc) {
    const float* xc = x + ((size_t)b * CCH + kc * 32 + kb) * NTOK + n;
    v16bf t;
    #pragma unroll
    for (int e = 0; e < 8; ++e) t[e]     = (__bf16)xc[(size_t)e * NTOK];
    #pragma unroll
    for (int e = 0; e < 8; ++e) t[8 + e] = (__bf16)xc[(size_t)(16 + e) * NTOK];
    bt[kc] = t;
  }

  // One 16-output-row GEMM tile over the full K=256 reduction (bf16 weights:
  // pure 16-byte loads, no per-wave conversion).
  auto do_tile = [&](const __bf16* W, const float* bias, int o0) -> v8f {
    v8f acc = {};
    #pragma unroll
    for (int kc = 0; kc < 8; ++kc) {
      const __bf16* wr = W + (size_t)(o0 + l16) * CCH + kc * 32 + kb;
      acc = wmma_bf16(load_op(wr), bt[kc], acc);
    }
    #pragma unroll
    for (int r = 0; r < 8; ++r) acc[r] += bias[o0 + kb + r];
    return acc;
  };

  // Q and K (32 output channels = 2 tiles each), stored transposed [n][o].
  #pragma unroll
  for (int ot = 0; ot < 2; ++ot) {
    v8f aq = do_tile(Wq, bq, ot * 16);
    v8bf p;
    #pragma unroll
    for (int r = 0; r < 8; ++r) p[r] = (__bf16)aq[r];
    *(v8bf*)(qT + ((size_t)b * NTOK + n) * CQK + ot * 16 + kb) = p;

    v8f ak = do_tile(Wk, bk, ot * 16);
    #pragma unroll
    for (int r = 0; r < 8; ++r) p[r] = (__bf16)ak[r];
    *(v8bf*)(kT + ((size_t)b * NTOK + n) * CQK + ot * 16 + kb) = p;
  }

  // V (256 output channels = 16 tiles), stored [c][n].
  #pragma unroll
  for (int ot = 0; ot < 16; ++ot) {
    v8f av = do_tile(Wv, bv, ot * 16);
    #pragma unroll
    for (int r = 0; r < 8; ++r) {
      const int c = ot * 16 + kb + r;
      vB[((size_t)b * CCH + c) * NTOK + n] = (__bf16)av[r];
    }
  }
}

// ---------------------------------------------------------------------------
// Kernel 2: fused flash attention. s[i,j] = q_i . k_j, softmax over i,
// o[c,j] = sum_i v[c,i] * p[i,j]; out = x + gamma * o / l.
// Block = 8 waves covering 64 j-columns: waves pair up per 16-j group and
// split the 256 V-channels in half (2x occupancy vs. full-channel waves at
// the cost of one duplicated S tile). V chunks are double-buffered in LDS
// via async global->LDS copies and shared by all 8 waves.
// ---------------------------------------------------------------------------
__global__ __launch_bounds__(256)
void attn_kernel(const float* __restrict__ x,
                 const __bf16* __restrict__ qT, const __bf16* __restrict__ kT,
                 const __bf16* __restrict__ vB,
                 const float* __restrict__ gamma,
                 float* __restrict__ out)
{
  __shared__ __bf16 ldsV[2][CCH * 32];       // 2 x 16 KB double buffer

  const int b      = blockIdx.x >> 6;        // 64 blocks per batch
  const int jblk   = blockIdx.x & 63;
  const int wave   = threadIdx.x >> 5;
  const int jgrp   = wave >> 1;              // 4 j-groups of 16 columns
  const int chalf  = wave & 1;               // low/high 128 channels
  const int lane   = threadIdx.x & 31;
  const int l16    = lane & 15;
  const int kb     = (lane >= 16) ? 8 : 0;
  const int j      = jblk * 64 + jgrp * 16 + l16;
  const int tid    = threadIdx.x;

  // Loop-invariant B operand: K-projection column j (the flash "query").
  const __bf16* kp = kT + ((size_t)b * NTOK + j) * CQK + kb;
  const v16bf bK = load_op(kp);

  v8f acc[8];
  #pragma unroll
  for (int t = 0; t < 8; ++t) acc[t] = (v8f){};
  float mrun = -INFINITY, lrun = 0.f;

  const __bf16* vbase = vB + (size_t)b * CCH * NTOK;

  auto stage = [&](int buf, int i0) {
    // 256 threads: thread t copies V row c=t, 32 i-values (64 B) to LDS.
    const __bf16* g = vbase + (size_t)tid * NTOK + i0;
#if USE_ASYNC_LDS
    unsigned lds0 = (unsigned)(uintptr_t)(&ldsV[buf][tid * 32]);
    unsigned long long ga = (unsigned long long)(uintptr_t)g;
    #pragma unroll
    for (int q = 0; q < 4; ++q) {
      asm volatile("global_load_async_to_lds_b128 %0, %1, off"
                   :: "v"(lds0 + q * 16), "v"(ga + (unsigned long long)(q * 16))
                   : "memory");
    }
#else
    #pragma unroll
    for (int q = 0; q < 4; ++q)
      *(v8bf*)(&ldsV[buf][tid * 32 + q * 8]) = *(const v8bf*)(g + q * 8);
#endif
  };

  stage(0, 0);

  for (int ch = 0; ch < NTOK / 32; ++ch) {
    const int i0  = ch * 32;
    const int cur = ch & 1;

#if USE_ASYNC_LDS
    asm volatile("s_wait_asynccnt 0x0" ::: "memory");
#endif
    __syncthreads();                              // staged buffer `cur` ready
    if (ch + 1 < NTOK / 32) stage(cur ^ 1, i0 + 32);

    // A operands: Q-projection rows i0..i0+15 and i0+16..i0+31.
    const __bf16* qp0 = qT + ((size_t)b * NTOK + i0 + l16) * CQK + kb;
    __builtin_prefetch(qp0 + 32 * CQK, 0, 0);     // next chunk's Q rows
    const v16bf a0 = load_op(qp0);
    const v16bf a1 = load_op(qp0 + 16 * CQK);

    v8f zc = {};
    v8f s0 = wmma_bf16(a0, bK, zc);               // s[i0..i0+15 , j-tile]
    v8f s1 = wmma_bf16(a1, bK, zc);               // s[i0+16..i0+31, j-tile]

    // Online softmax over i for each column j (lanes j and j+16 pair up).
    float cm = s0[0];
    #pragma unroll
    for (int r = 1; r < 8; ++r) cm = fmaxf(cm, s0[r]);
    #pragma unroll
    for (int r = 0; r < 8; ++r) cm = fmaxf(cm, s1[r]);
    cm = fmaxf(cm, __shfl_xor(cm, 16, 32));
    const float mnew  = fmaxf(mrun, cm);
    const float scale = __expf(mrun - mnew);

    // S D-layout maps directly onto P's 32x16 B-operand layout: no shuffles.
    v16bf P;
    float rs = 0.f;
    #pragma unroll
    for (int r = 0; r < 8; ++r) { float e = __expf(s0[r] - mnew); rs += e; P[r]     = (__bf16)e; }
    #pragma unroll
    for (int r = 0; r < 8; ++r) { float e = __expf(s1[r] - mnew); rs += e; P[8 + r] = (__bf16)e; }
    lrun = lrun * scale + rs + __shfl_xor(rs, 16, 32);
    mrun = mnew;

    #pragma unroll
    for (int t = 0; t < 8; ++t) {
      #pragma unroll
      for (int r = 0; r < 8; ++r) acc[t][r] *= scale;
    }

    // O += V * P over this wave's 128 channels, V tiles from shared LDS.
    const __bf16* lv = &ldsV[cur][0];
    #pragma unroll
    for (int t = 0; t < 8; ++t) {
      const int c0 = chalf * 128 + t * 16;
      const __bf16* vp = lv + (c0 + l16) * 32 + kb;
      const v16bf av = load_op(vp);
      acc[t] = wmma_bf16(av, P, acc[t]);
    }
  }

  // Epilogue: normalize, residual add, write f32 output.
  const float g    = gamma[0];
  const float invl = 1.0f / lrun;
  #pragma unroll
  for (int t = 0; t < 8; ++t) {
    #pragma unroll
    for (int r = 0; r < 8; ++r) {
      const int c = chalf * 128 + t * 16 + kb + r;
      const size_t idx = ((size_t)b * CCH + c) * NTOK + j;
      out[idx] = x[idx] + g * (acc[t][r] * invl);
    }
  }
}

extern "C" void kernel_launch(void* const* d_in, const int* in_sizes, int n_in,
                              void* d_out, int out_size, void* d_ws, size_t ws_size,
                              hipStream_t stream) {
  const float* x  = (const float*)d_in[0];
  const float* Wq = (const float*)d_in[1];
  const float* bq = (const float*)d_in[2];
  const float* Wk = (const float*)d_in[3];
  const float* bk = (const float*)d_in[4];
  const float* Wv = (const float*)d_in[5];
  const float* bv = (const float*)d_in[6];
  const float* gm = (const float*)d_in[7];

  __bf16* qT  = (__bf16*)d_ws;                      // 4*4096*32 bf16 = 1 MB
  __bf16* kT  = qT + (size_t)NBAT * NTOK * CQK;     // 1 MB
  __bf16* vB  = kT + (size_t)NBAT * NTOK * CQK;     // 4*256*4096 bf16 = 8 MB
  __bf16* wqb = vB + (size_t)NBAT * CCH * NTOK;     // 16 KB
  __bf16* wkb = wqb + (size_t)CQK * CCH;            // 16 KB
  __bf16* wvb = wkb + (size_t)CQK * CCH;            // 128 KB
  float* out = (float*)d_out;

  cvt_weights<<<dim3((CCH * CCH + 255) / 256), dim3(256), 0, stream>>>(
      Wq, Wk, Wv, wqb, wkb, wvb);
  proj_kernel<<<dim3(NBAT * 32), dim3(256), 0, stream>>>(
      x, wqb, bq, wkb, bk, wvb, bv, qT, kT, vB);
  attn_kernel<<<dim3(NBAT * 64), dim3(256), 0, stream>>>(
      x, qT, kT, vB, gm, out);
}
